// GraphAttention_1872605741508
// MI455X (gfx1250) — matrix operand order repaired and checked by reading, not compiled, not run on hardware
//
#include <hip/hip_runtime.h>
#include <hip/hip_bf16.h>

// ---------------------------------------------------------------------------
// GAT forward for MI455X (gfx1250, wave32).
//   feats = X @ W          -> v_wmma_f32_16x16x32_f16 (f32 accumulate)
//   sparse softmax + SpMM  -> L2-resident gathers + f32 global atomics
// ---------------------------------------------------------------------------

typedef _Float16 v16h __attribute__((ext_vector_type(16)));
typedef _Float16 v8h  __attribute__((ext_vector_type(8)));
typedef float    v8f  __attribute__((ext_vector_type(8)));

#define N_NODES 50000
#define N_EDGES 1600000
#define F_IN    256
#define F_OUT   64
#define HEADS   4
#define COLS    256   // HEADS * F_OUT

// ---- orderable-uint encoding of float for atomicMax --------------------------
__device__ __forceinline__ unsigned fenc(float f) {
    unsigned u = __float_as_uint(f);
    return (u & 0x80000000u) ? ~u : (u | 0x80000000u);
}
__device__ __forceinline__ float fdec(unsigned u) {
    return (u & 0x80000000u) ? __uint_as_float(u & 0x7FFFFFFFu)
                             : __uint_as_float(~u);
}
__device__ __forceinline__ float leaky(float x) { return x > 0.f ? x : 0.2f * x; }

// ---- init: zero output + denom, -inf segmax ---------------------------------
__global__ void k_init(float* __restrict__ out, unsigned* __restrict__ segmax,
                       float* __restrict__ denom) {
    int i = blockIdx.x * blockDim.x + threadIdx.x;
    if (i < N_NODES * COLS) out[i] = 0.f;
    if (i < N_NODES * HEADS) { segmax[i] = 0u; denom[i] = 0.f; }
}

// ---- pack W [H,F_IN,F_OUT] f32 -> Bmat f16, column-major per output col -----
// Bmat[col*256 + k] = (f16) W[h, k, o]  with col = h*64 + o
__global__ void k_build_b(const float* __restrict__ W, _Float16* __restrict__ Bmat) {
    int idx = blockIdx.x * blockDim.x + threadIdx.x;   // 65536 threads
    int col = idx >> 8, k = idx & 255;
    int h = col >> 6, o = col & 63;
    Bmat[col * F_IN + k] = (_Float16)W[(h * F_IN + k) * F_OUT + o];
}

// ---- WMMA GEMM: feats[N][256] = X[N][256] @ Bmat ----------------------------
// One wave per 16x16 output tile; 50000 tiles = 6250 blocks * 8 waves exactly.
__global__ void __launch_bounds__(256)
k_gemm(const float* __restrict__ X, const _Float16* __restrict__ Bmat,
       float* __restrict__ feats) {
    const int lane = threadIdx.x & 31;
    const int wid  = blockIdx.x * 8 + (threadIdx.x >> 5);
    const int tm   = wid >> 4;          // 0..3124  (row tile)
    const int tn   = wid & 15;          // 0..15    (col tile)
    const int half = lane >> 4;         // K-half selector
    const int m    = lane & 15;         // row (A) / col (B,C)

    const float*    xrow = X    + (size_t)(tm * 16 + m) * F_IN;
    const _Float16* bcol = Bmat + (size_t)(tn * 16 + m) * F_IN;

    v8f c = {};
#pragma unroll
    for (int kb = 0; kb < 256; kb += 32) {
        // A fragment: 16x32 f16; lane half h: elems 0..7 = K kb+h*16..,
        //                                    elems 8..15 = K kb+h*16+8..
        const float* pa = xrow + kb + half * 16;
        const float* pb = xrow + kb + half * 16 + 8;
        v16h a;
#pragma unroll
        for (int i = 0; i < 8; ++i) { a[i] = (_Float16)pa[i]; a[8 + i] = (_Float16)pb[i]; }

        // B fragment: 32x16 f16; lane half h holds K = kb + h*16 + {0..15},
        // contiguous in our column-major Bmat -> two 16B loads.
        const v8h* bp = (const v8h*)(bcol + kb + half * 16);
        v8h b0 = bp[0], b1 = bp[1];
        v16h b;
#pragma unroll
        for (int i = 0; i < 8; ++i) { b[i] = b0[i]; b[8 + i] = b1[i]; }

        c = __builtin_amdgcn_wmma_f32_16x16x32_f16(
                /*neg_a=*/false, a, /*neg_b=*/false, b,
                /*c_mod=*/(short)0, c, /*reuse_a=*/false, /*reuse_b=*/false);
    }

    // C/D: VGPR v -> row tm*16 + half*8 + v, col tn*16 + m
    float* frow = feats + (size_t)(tm * 16 + half * 8) * COLS + tn * 16 + m;
#pragma unroll
    for (int v = 0; v < 8; ++v) frow[(size_t)v * COLS] = c[v];
}

// ---- attention logits: a_s/a_n [N][H] ---------------------------------------
__global__ void k_logits(const float* __restrict__ feats,
                         const float* __restrict__ ak_s,
                         const float* __restrict__ ak_n,
                         float* __restrict__ a_s, float* __restrict__ a_n) {
    int t = blockIdx.x * blockDim.x + threadIdx.x;
    if (t >= N_NODES * HEADS) return;
    int n = t >> 2, h = t & 3;
    const float* f = feats + (size_t)n * COLS + h * F_OUT;
    float ss = 0.f, sn = 0.f;
#pragma unroll 8
    for (int o = 0; o < F_OUT; ++o) {
        float v = f[o];
        ss += v * ak_s[h * F_OUT + o];
        sn += v * ak_n[h * F_OUT + o];
    }
    a_s[n * HEADS + h] = ss;
    a_n[n * HEADS + h] = sn;
}

// ---- pass 1: per-row segment max of leaky scores ----------------------------
__global__ void k_edge_max(const int* __restrict__ rows, const int* __restrict__ cols,
                           const float* __restrict__ adj,
                           const float* __restrict__ a_s, const float* __restrict__ a_n,
                           unsigned* __restrict__ segmax) {
    int e = blockIdx.x * blockDim.x + threadIdx.x;
    if (e >= N_EDGES) return;
    int r = rows[e], c = cols[e];
    float av = adj[e];
#pragma unroll
    for (int h = 0; h < HEADS; ++h) {
        float sc = leaky(av * (a_s[r * HEADS + h] + a_n[c * HEADS + h]));
        atomicMax(&segmax[r * HEADS + h], fenc(sc));
    }
}

// ---- pass 2: exp + denominator ----------------------------------------------
__global__ void k_edge_sum(const int* __restrict__ rows, const int* __restrict__ cols,
                           const float* __restrict__ adj,
                           const float* __restrict__ a_s, const float* __restrict__ a_n,
                           const unsigned* __restrict__ segmax,
                           float* __restrict__ denom) {
    int e = blockIdx.x * blockDim.x + threadIdx.x;
    if (e >= N_EDGES) return;
    int r = rows[e], c = cols[e];
    float av = adj[e];
#pragma unroll
    for (int h = 0; h < HEADS; ++h) {
        float sc  = leaky(av * (a_s[r * HEADS + h] + a_n[c * HEADS + h]));
        float att = __expf(sc - fdec(segmax[r * HEADS + h]));
        atomicAdd(&denom[r * HEADS + h], att);
    }
}

// ---- pass 3: wave-per-edge SpMM aggregation ---------------------------------
// lane l covers output cols [l*8, l*8+8) of 256; its head = l>>3.
// feats row of the source node is L2-resident (51 MB < 192 MB L2).
__global__ void __launch_bounds__(256)
k_aggregate(const int* __restrict__ rows, const int* __restrict__ cols,
            const float* __restrict__ adj,
            const float* __restrict__ a_s, const float* __restrict__ a_n,
            const unsigned* __restrict__ segmax, const float* __restrict__ denom,
            const float* __restrict__ feats, float* __restrict__ out) {
    const int lane = threadIdx.x & 31;
    const int e    = blockIdx.x * 8 + (threadIdx.x >> 5);   // 200000 blocks exact
    const int r = rows[e], c = cols[e];
    const int h = lane >> 3;

    float sc  = leaky(adj[e] * (a_s[r * HEADS + h] + a_n[c * HEADS + h]));
    float att = __expf(sc - fdec(segmax[r * HEADS + h])) / denom[r * HEADS + h];

    const float4* f4 = (const float4*)(feats + (size_t)c * COLS + lane * 8);
    float4 f0 = f4[0], f1 = f4[1];

    float* o = out + (size_t)r * COLS + lane * 8;
    atomicAdd(o + 0, att * f0.x);
    atomicAdd(o + 1, att * f0.y);
    atomicAdd(o + 2, att * f0.z);
    atomicAdd(o + 3, att * f0.w);
    atomicAdd(o + 4, att * f1.x);
    atomicAdd(o + 5, att * f1.y);
    atomicAdd(o + 6, att * f1.z);
    atomicAdd(o + 7, att * f1.w);
}

// ---- finalize: bias + relu ---------------------------------------------------
__global__ void k_final(float* __restrict__ out, const float* __restrict__ bias) {
    int i = blockIdx.x * blockDim.x + threadIdx.x;
    if (i >= N_NODES * COLS) return;
    float v = out[i] + bias[i & 255];   // bias flat index == output column
    out[i] = v > 0.f ? v : 0.f;
}

// ---------------------------------------------------------------------------
extern "C" void kernel_launch(void* const* d_in, const int* in_sizes, int n_in,
                              void* d_out, int out_size, void* d_ws, size_t ws_size,
                              hipStream_t stream) {
    const float* X    = (const float*)d_in[0];
    const float* adjv = (const float*)d_in[1];
    const float* W    = (const float*)d_in[2];
    const float* bias = (const float*)d_in[3];
    const float* ak_s = (const float*)d_in[4];
    const float* ak_n = (const float*)d_in[5];
    const int*   rows = (const int*)d_in[6];
    const int*   cols = (const int*)d_in[7];
    float* out = (float*)d_out;

    // workspace carve-up (all 256B-aligned)
    char* w = (char*)d_ws;
    auto take = [&](size_t bytes) {
        char* p = w;
        w += (bytes + 255) & ~(size_t)255;
        return p;
    };
    float*    feats  = (float*)   take((size_t)N_NODES * COLS * 4);   // 51.2 MB
    _Float16* Bmat   = (_Float16*)take((size_t)COLS * F_IN * 2);      // 128 KB
    float*    a_s    = (float*)   take((size_t)N_NODES * HEADS * 4);
    float*    a_n    = (float*)   take((size_t)N_NODES * HEADS * 4);
    unsigned* segmax = (unsigned*)take((size_t)N_NODES * HEADS * 4);
    float*    denom  = (float*)   take((size_t)N_NODES * HEADS * 4);

    const int T = 256;
    k_init   <<<(N_NODES * COLS + T - 1) / T, T, 0, stream>>>(out, segmax, denom);
    k_build_b<<<(COLS * F_IN) / T, T, 0, stream>>>(W, Bmat);
    k_gemm   <<<(N_NODES / 16) * 16 / 8, T, 0, stream>>>(X, Bmat, feats); // 6250 blocks
    k_logits <<<(N_NODES * HEADS + T - 1) / T, T, 0, stream>>>(feats, ak_s, ak_n, a_s, a_n);
    k_edge_max<<<(N_EDGES + T - 1) / T, T, 0, stream>>>(rows, cols, adjv, a_s, a_n, segmax);
    k_edge_sum<<<(N_EDGES + T - 1) / T, T, 0, stream>>>(rows, cols, adjv, a_s, a_n, segmax, denom);
    k_aggregate<<<N_EDGES / 8, T, 0, stream>>>(rows, cols, adjv, a_s, a_n, segmax, denom, feats, out);
    k_final  <<<(N_NODES * COLS + T - 1) / T, T, 0, stream>>>(out, bias);
}
